// WindowAttention_sparse_70308614635682
// MI455X (gfx1250) — matrix operand order repaired and checked by reading, not compiled, not run on hardware
//
#include <hip/hip_runtime.h>
#include <hip/hip_bf16.h>
#include <stdint.h>

typedef __attribute__((ext_vector_type(16))) __bf16 bf16x16;
typedef __attribute__((ext_vector_type(8)))  float  f32x8;

#define DIM      512
#define HEADS    16
#define HEAD_DIM 32
#define NTOK     64
#define OUTC     1536
#define SCALE_F  0.17677669529663687f   // 32^-0.5

// ---- WMMA fragment index helpers (CDNA5 ISA §7.12.2) ----
// A-matrix 16x32 bf16: lane = 16*half + m. element e -> K index:
//   half=0: K = 0..7, 16..23   half=1: K = 8..15, 24..31
__device__ __forceinline__ int a_kidx(int e, int half) {
  return e + (e < 8 ? (half ? 8 : 0) : (half ? 16 : 8));
}
// B as loaded in this file (A too): lane = 16*half + n. element e -> K = e + 16*half.

// ---- CDNA5 async global->LDS copy (16B per lane), tracked by ASYNCcnt ----
__device__ __forceinline__ void lds_async_b128(void* lds_dst, const void* gsrc) {
  uint32_t l = (uint32_t)(uintptr_t)lds_dst;   // low 32 bits = LDS offset
  asm volatile("global_load_async_to_lds_b128 %0, %1, off"
               :: "v"(l), "v"(gsrc) : "memory");
}
__device__ __forceinline__ void wait_async0() {
  asm volatile("s_wait_asynccnt 0" ::: "memory");
}

// =====================================================================
// Kernel 1: weights -> bf16.  Wq(512x512)+Wkv(512x1024) fused to 512x1536.
// =====================================================================
__global__ void wa_wconvert(const float* __restrict__ Wq,
                            const float* __restrict__ Wkv,
                            const float* __restrict__ Wp,
                            __bf16* __restrict__ wc,
                            __bf16* __restrict__ wpb) {
  int idx = blockIdx.x * blockDim.x + threadIdx.x;
  const int total1 = DIM * OUTC;
  if (idx < total1) {
    int k = idx / OUTC, j = idx - k * OUTC;
    float v = (j < DIM) ? Wq[(size_t)k * DIM + j]
                        : Wkv[(size_t)k * (2 * DIM) + (j - DIM)];
    wc[idx] = (__bf16)v;
  } else {
    int i2 = idx - total1;
    if (i2 < DIM * DIM) wpb[i2] = (__bf16)Wp[i2];
  }
}

// =====================================================================
// Kernel 2: fused QKV projection.
// block = (window b, 64-col slab).  128 threads = 4 waves.
// =====================================================================
__global__ __launch_bounds__(128)
void wa_qkv_gemm(const float* __restrict__ x, const __bf16* __restrict__ wc,
                 const float* __restrict__ bq, const float* __restrict__ bkv,
                 __bf16* __restrict__ qb, __bf16* __restrict__ kb,
                 __bf16* __restrict__ vb) {
  const int b    = blockIdx.y;
  const int n0   = blockIdx.x * 64;
  const int tid  = threadIdx.x;
  const int wv   = tid >> 5;
  const int lane = tid & 31;
  const int half = lane >> 4;
  const int lm   = lane & 15;

  __shared__ __bf16 xs[64][34];   // [m][k_local]  (A tile: transpose+convert)
  __shared__ __bf16 wt[64][34];   // [n][k_local]  (B tile: transposed)

  f32x8 acc[4];
#pragma unroll
  for (int t = 0; t < 4; ++t) acc[t] = (f32x8){0.f,0.f,0.f,0.f,0.f,0.f,0.f,0.f};

  const float* xb = x + (size_t)b * DIM * NTOK;
  const int kl  = tid >> 2;          // 0..31
  const int c16 = (tid & 3) << 4;    // 0,16,32,48

  for (int kk = 0; kk < DIM; kk += 32) {
    {  // stage A: x[b, kk+kl, c16..+15] -> xs[m][kl]
      const float4* src = (const float4*)(xb + (size_t)(kk + kl) * NTOK + c16);
      float4 v0 = src[0], v1 = src[1], v2 = src[2], v3 = src[3];
      float t16[16] = {v0.x,v0.y,v0.z,v0.w, v1.x,v1.y,v1.z,v1.w,
                       v2.x,v2.y,v2.z,v2.w, v3.x,v3.y,v3.z,v3.w};
#pragma unroll
      for (int i = 0; i < 16; ++i) xs[c16 + i][kl] = (__bf16)t16[i];
    }
    {  // stage B: wc[kk+kl, n0+c16..+15] -> wt[n][kl]
      const __bf16* wsrc = wc + (size_t)(kk + kl) * OUTC + n0 + c16;
#pragma unroll
      for (int i = 0; i < 16; ++i) wt[c16 + i][kl] = wsrc[i];
    }
    __syncthreads();

    if (kk + 32 < DIM) {  // prefetch next K tile
      __builtin_prefetch(xb + (size_t)(kk + 32 + kl) * NTOK + c16, 0, 0);
      __builtin_prefetch(wc + (size_t)(kk + 32 + kl) * OUTC + n0 + c16, 0, 0);
    }

    bf16x16 afrag;
    const __bf16* arow = &xs[wv * 16 + lm][0];
#pragma unroll
    for (int e = 0; e < 16; ++e) afrag[e] = arow[e + 16 * half];

#pragma unroll
    for (int nt = 0; nt < 4; ++nt) {
      bf16x16 bfrag;
      const __bf16* brow = &wt[nt * 16 + lm][0];
#pragma unroll
      for (int e = 0; e < 16; ++e) bfrag[e] = brow[e + 16 * half];
      acc[nt] = __builtin_amdgcn_wmma_f32_16x16x32_bf16(
          false, afrag, false, bfrag, (short)0, acc[nt], false, false);
    }
    __syncthreads();
  }

  // epilogue: +bias, scatter into q/k/v (B,H,64,32) bf16
#pragma unroll
  for (int nt = 0; nt < 4; ++nt) {
    int j = n0 + nt * 16 + lm;        // 0..1535
    float bias; __bf16* dst; int jj;
    if (j < DIM)            { bias = bq[j];        jj = j;           dst = qb; }
    else if (j < 2 * DIM)   { bias = bkv[j - DIM]; jj = j - DIM;     dst = kb; }
    else                    { bias = bkv[j - DIM]; jj = j - 2 * DIM; dst = vb; }
    int h = jj >> 5, d = jj & 31;
    size_t base = (((size_t)b * HEADS + h) * NTOK) * HEAD_DIM + d;
#pragma unroll
    for (int r = 0; r < 8; ++r) {
      int m = wv * 16 + half * 8 + r;
      dst[base + (size_t)m * HEAD_DIM] = (__bf16)(acc[nt][r] + bias);
    }
  }
}

// =====================================================================
// Kernel 3: attention core per (b,h).  256 threads = 8 waves.
// q/k staged by async global->LDS DMA; v transposed manually.
// =====================================================================
__global__ __launch_bounds__(256)
void wa_attn(const __bf16* __restrict__ qb, const __bf16* __restrict__ kb,
             const __bf16* __restrict__ vb, const float* __restrict__ bias_table,
             const float* __restrict__ w_blend, __bf16* __restrict__ ob) {
  const int b = blockIdx.y, h = blockIdx.x;
  const int tid  = threadIdx.x;
  const int wv   = tid >> 5;
  const int lane = tid & 31;
  const int half = lane >> 4;
  const int lm   = lane & 15;

  __shared__ float  Sf[64][66];     // logits (fp32)
  __shared__ __bf16 Ab[64][66];     // blended probs (bf16)
  __shared__ __bf16 qs[64][40];     // q  [m][d]   (pitch 80B: 16B aligned rows)
  __shared__ __bf16 ks[64][40];     // k  [n][d]
  __shared__ __bf16 vt[32][66];     // vT [d][m]
  __shared__ float  bh[225];        // this head's bias table

  const size_t base = (((size_t)b * HEADS + h) * NTOK) * HEAD_DIM;
  {
    int m  = tid >> 2;              // 0..63
    int d0 = (tid & 3) << 3;        // 0,8,16,24
    // async DMA copies (straight, 16B each) for q and k
    lds_async_b128(&qs[m][d0], qb + base + (size_t)m * HEAD_DIM + d0);
    lds_async_b128(&ks[m][d0], kb + base + (size_t)m * HEAD_DIM + d0);
    // manual transpose for v
    const __bf16* vsrc = vb + base + (size_t)m * HEAD_DIM + d0;
#pragma unroll
    for (int i = 0; i < 8; ++i) vt[d0 + i][m] = vsrc[i];
  }
  if (tid < 225) bh[tid] = bias_table[tid * HEADS + h];
  wait_async0();
  __syncthreads();

  // blend weights = softmax(w_blend)
  float wb0 = w_blend[0], wb1 = w_blend[1];
  float wmx = fmaxf(wb0, wb1);
  float ea = __expf(wb0 - wmx), eb = __expf(wb1 - wmx);
  float w0 = ea / (ea + eb), w1 = eb / (ea + eb);

  // ---- S = (q @ k^T) * SCALE + bias : 16 tiles, 2 per wave (K=32 -> 1 WMMA)
#pragma unroll
  for (int t = 0; t < 2; ++t) {
    int tile = wv * 2 + t;
    int mt = tile >> 2, nt = tile & 3;
    bf16x16 afrag, bfrag;
    const __bf16* arow = &qs[mt * 16 + lm][0];
    const __bf16* brow = &ks[nt * 16 + lm][0];
#pragma unroll
    for (int e = 0; e < 16; ++e) {
      afrag[e] = arow[e + 16 * half];
      bfrag[e] = brow[e + 16 * half];
    }
    f32x8 acc = (f32x8){0.f,0.f,0.f,0.f,0.f,0.f,0.f,0.f};
    acc = __builtin_amdgcn_wmma_f32_16x16x32_bf16(
        false, afrag, false, bfrag, (short)0, acc, false, false);
#pragma unroll
    for (int r = 0; r < 8; ++r) {
      int i = mt * 16 + half * 8 + r;      // query token
      int j = nt * 16 + lm;                // key token
      int yi = i >> 3, xi = i & 7, yj = j >> 3, xj = j & 7;
      float bias = bh[(yi - yj + 7) * 15 + (xi - xj + 7)];
      Sf[i][j] = acc[r] * SCALE_F + bias;
    }
  }
  __syncthreads();

  // ---- softmax + relu^2 blend: 4 lanes per row, shfl_xor reductions
  {
    int row = tid >> 2;
    int c0  = (tid & 3) << 4;
    float lmax = -1e30f;
#pragma unroll
    for (int j = 0; j < 16; ++j) lmax = fmaxf(lmax, Sf[row][c0 + j]);
    lmax = fmaxf(lmax, __shfl_xor(lmax, 1, 32));
    lmax = fmaxf(lmax, __shfl_xor(lmax, 2, 32));
    float lsum = 0.f;
#pragma unroll
    for (int j = 0; j < 16; ++j) lsum += __expf(Sf[row][c0 + j] - lmax);
    lsum += __shfl_xor(lsum, 1, 32);
    lsum += __shfl_xor(lsum, 2, 32);
    float inv = 1.f / lsum;
#pragma unroll
    for (int j = 0; j < 16; ++j) {
      float s  = Sf[row][c0 + j];
      float p  = __expf(s - lmax) * inv;
      float rl = fmaxf(s, 0.f);
      Ab[row][c0 + j] = (__bf16)(w0 * p + w1 * rl * rl);
    }
  }
  __syncthreads();

  // ---- O = A @ V : 8 tiles (4x2), one per wave, K = 64 (two WMMAs)
  {
    int mt = wv >> 1, nt = wv & 1;
    f32x8 acc = (f32x8){0.f,0.f,0.f,0.f,0.f,0.f,0.f,0.f};
#pragma unroll
    for (int kk = 0; kk < 64; kk += 32) {
      bf16x16 afrag, bfrag;
      const __bf16* arow = &Ab[mt * 16 + lm][kk];
      const __bf16* brow = &vt[nt * 16 + lm][kk];
#pragma unroll
      for (int e = 0; e < 16; ++e) {
        afrag[e] = arow[e + 16 * half];
        bfrag[e] = brow[e + 16 * half];
      }
      acc = __builtin_amdgcn_wmma_f32_16x16x32_bf16(
          false, afrag, false, bfrag, (short)0, acc, false, false);
    }
#pragma unroll
    for (int r = 0; r < 8; ++r) {
      int m = mt * 16 + half * 8 + r;
      int d = nt * 16 + lm;
      ob[((size_t)b * NTOK + m) * DIM + h * HEAD_DIM + d] = (__bf16)acc[r];
    }
  }
}

// =====================================================================
// Kernel 4: output projection + transpose to (B, 512, 64) fp32.
// A tile staged by async global->LDS DMA (straight bf16 copy).
// =====================================================================
__global__ __launch_bounds__(128)
void wa_proj_gemm(const __bf16* __restrict__ ob, const __bf16* __restrict__ wpb,
                  const float* __restrict__ bp, float* __restrict__ out) {
  const int b    = blockIdx.y;
  const int n0   = blockIdx.x * 64;
  const int tid  = threadIdx.x;
  const int wv   = tid >> 5;
  const int lane = tid & 31;
  const int half = lane >> 4;
  const int lm   = lane & 15;

  __shared__ __bf16 os[64][40];   // [m][k_local] (pitch 80B: 16B aligned rows)
  __shared__ __bf16 wt[64][34];   // [n][k_local]

  f32x8 acc[4];
#pragma unroll
  for (int t = 0; t < 4; ++t) acc[t] = (f32x8){0.f,0.f,0.f,0.f,0.f,0.f,0.f,0.f};

  const int kl  = tid >> 2;             // 0..31
  const int c16 = (tid & 3) << 4;       // 0,16,32,48
  const int am  = tid >> 1;             // 0..63
  const int ak0 = (tid & 1) << 4;       // 0,16

  for (int kk = 0; kk < DIM; kk += 32) {
    {  // stage A via async DMA: ob[b, am, kk+ak0 .. +15]
      const __bf16* s = ob + ((size_t)b * NTOK + am) * DIM + kk + ak0;
      lds_async_b128(&os[am][ak0],     s);
      lds_async_b128(&os[am][ak0 + 8], s + 8);
    }
    {  // stage B transposed
      const __bf16* wsrc = wpb + (size_t)(kk + kl) * DIM + n0 + c16;
#pragma unroll
      for (int i = 0; i < 16; ++i) wt[c16 + i][kl] = wsrc[i];
    }
    wait_async0();
    __syncthreads();

    if (kk + 32 < DIM) {
      __builtin_prefetch(ob + ((size_t)b * NTOK + am) * DIM + kk + 32 + ak0, 0, 0);
      __builtin_prefetch(wpb + (size_t)(kk + 32 + kl) * DIM + n0 + c16, 0, 0);
    }

    bf16x16 afrag;
    const __bf16* arow = &os[wv * 16 + lm][0];
#pragma unroll
    for (int e = 0; e < 16; ++e) afrag[e] = arow[e + 16 * half];

#pragma unroll
    for (int nt = 0; nt < 4; ++nt) {
      bf16x16 bfrag;
      const __bf16* brow = &wt[nt * 16 + lm][0];
#pragma unroll
      for (int e = 0; e < 16; ++e) bfrag[e] = brow[e + 16 * half];
      acc[nt] = __builtin_amdgcn_wmma_f32_16x16x32_bf16(
          false, afrag, false, bfrag, (short)0, acc[nt], false, false);
    }
    __syncthreads();
  }

  // epilogue: +bp, write transposed (B, C, N) fp32 — 8 contiguous floats/lane
#pragma unroll
  for (int nt = 0; nt < 4; ++nt) {
    int n = n0 + nt * 16 + lm;
    float bias = bp[n];
    float* dst = out + (size_t)b * DIM * NTOK + (size_t)n * NTOK + wv * 16 + half * 8;
    float4 lo = make_float4(acc[nt][0] + bias, acc[nt][1] + bias,
                            acc[nt][2] + bias, acc[nt][3] + bias);
    float4 hi = make_float4(acc[nt][4] + bias, acc[nt][5] + bias,
                            acc[nt][6] + bias, acc[nt][7] + bias);
    *(float4*)(dst)     = lo;
    *(float4*)(dst + 4) = hi;
  }
}

// =====================================================================
extern "C" void kernel_launch(void* const* d_in, const int* in_sizes, int n_in,
                              void* d_out, int out_size, void* d_ws, size_t ws_size,
                              hipStream_t stream) {
  (void)n_in; (void)out_size; (void)ws_size;
  const float* x   = (const float*)d_in[0];
  const float* Wq  = (const float*)d_in[1];
  const float* bq  = (const float*)d_in[2];
  const float* Wkv = (const float*)d_in[3];
  const float* bkv = (const float*)d_in[4];
  const float* bt  = (const float*)d_in[5];
  const float* wbl = (const float*)d_in[6];
  const float* Wp  = (const float*)d_in[7];
  const float* bp  = (const float*)d_in[8];
  float* out = (float*)d_out;

  const int B = in_sizes[0] / (DIM * NTOK);   // 1024 windows

  char* ws = (char*)d_ws;
  __bf16* wc  = (__bf16*)ws;  ws += (size_t)DIM * OUTC * sizeof(__bf16);
  __bf16* wpb = (__bf16*)ws;  ws += (size_t)DIM * DIM  * sizeof(__bf16);
  const size_t qkv_elems = (size_t)B * HEADS * NTOK * HEAD_DIM;
  __bf16* qb = (__bf16*)ws;   ws += qkv_elems * sizeof(__bf16);
  __bf16* kb = (__bf16*)ws;   ws += qkv_elems * sizeof(__bf16);
  __bf16* vb = (__bf16*)ws;   ws += qkv_elems * sizeof(__bf16);
  __bf16* ob = (__bf16*)ws;   ws += (size_t)B * NTOK * DIM * sizeof(__bf16);

  {
    int total = DIM * OUTC + DIM * DIM;
    wa_wconvert<<<(total + 255) / 256, 256, 0, stream>>>(Wq, Wkv, Wp, wc, wpb);
  }
  wa_qkv_gemm<<<dim3(OUTC / 64, B), 128, 0, stream>>>(x, wc, bq, bkv, qb, kb, vb);
  wa_attn<<<dim3(HEADS, B), 256, 0, stream>>>(qb, kb, vb, bt, wbl, ob);
  wa_proj_gemm<<<dim3(DIM / 64, B), 128, 0, stream>>>(ob, wpb, bp, out);
}
